// MultiHeadAttention_807453852144
// MI455X (gfx1250) — matrix operand order repaired and checked by reading, not compiled, not run on hardware
//
#include <hip/hip_runtime.h>
#include <hip/hip_bf16.h>

// Problem constants (match reference)
#define BB  4
#define LL  2048
#define DD  1024
#define HH  16
#define DKK 64

typedef __attribute__((ext_vector_type(16))) __bf16 v16bf;
typedef __attribute__((ext_vector_type(8)))  __bf16 v8bf;
typedef __attribute__((ext_vector_type(8)))  float  v8f;

static __device__ __forceinline__ v16bf cat8(v8bf lo, v8bf hi) {
  return __builtin_shufflevector(lo, hi, 0,1,2,3,4,5,6,7,8,9,10,11,12,13,14,15);
}

// ---------------------------------------------------------------------------
// f32 -> bf16 elementwise cast (for weights)
// ---------------------------------------------------------------------------
__global__ __launch_bounds__(256) void cast_kernel(const float* __restrict__ x,
                                                   unsigned short* __restrict__ yu,
                                                   int n) {
  __bf16* y = reinterpret_cast<__bf16*>(yu);
  int i = blockIdx.x * blockDim.x + threadIdx.x;
  if (i < n) y[i] = (__bf16)x[i];
}

// ---------------------------------------------------------------------------
// LayerNorm over D=1024 per token, output bf16. One block per token.
// ---------------------------------------------------------------------------
__global__ __launch_bounds__(256) void ln_cast_kernel(const float* __restrict__ x,
                                                      const float* __restrict__ g,
                                                      const float* __restrict__ beta,
                                                      unsigned short* __restrict__ yu) {
  __bf16* y = reinterpret_cast<__bf16*>(yu);
  const int tok = blockIdx.x;
  const int tid = threadIdx.x;
  const float* row = x + (size_t)tok * DD;

  float v[4], s = 0.f, sq = 0.f;
#pragma unroll
  for (int i = 0; i < 4; ++i) {
    v[i] = row[tid + i * 256];
    s += v[i];
    sq += v[i] * v[i];
  }
#pragma unroll
  for (int m = 16; m >= 1; m >>= 1) {
    s  += __shfl_xor(s,  m, 32);
    sq += __shfl_xor(sq, m, 32);
  }
  __shared__ float reds[8], redq[8];
  const int wid = tid >> 5, lane = tid & 31;
  if (lane == 0) { reds[wid] = s; redq[wid] = sq; }
  __syncthreads();
  s = 0.f; sq = 0.f;
#pragma unroll
  for (int wi = 0; wi < 8; ++wi) { s += reds[wi]; sq += redq[wi]; }

  const float mu  = s * (1.0f / DD);
  const float var = sq * (1.0f / DD) - mu * mu;
  const float inv = rsqrtf(var + 1e-5f);

  __bf16* orow = y + (size_t)tok * DD;
#pragma unroll
  for (int i = 0; i < 4; ++i) {
    const int c = tid + i * 256;
    orow[c] = (__bf16)((v[i] - mu) * inv * g[c] + beta[c]);
  }
}

// ---------------------------------------------------------------------------
// Projection GEMM: out[n,j] = sum_d X[n,d] * W[j,d] + bias[j]
// Block: 256 threads = 8 waves; block tile 256 rows x 64 cols.
// Wave tile: 32 rows x 64 cols (2 A fragments, 8 accumulators).
// Weight K-slab staged in double-buffered LDS (2 x 64x32 bf16), software
// pipelined: next slab's global load is issued a full iteration early, so the
// ds_store never waits on an in-flight load. One barrier per K-step.
// transposeV==0 -> store (B,H,L,DK) bf16 ; transposeV==1 -> store (B,H,DK,L).
// ---------------------------------------------------------------------------
__global__ __launch_bounds__(256) void proj_kernel(const unsigned short* __restrict__ Xu,
                                                   const unsigned short* __restrict__ Wu,
                                                   const float* __restrict__ bias,
                                                   unsigned short* __restrict__ outu,
                                                   int transposeV) {
  const __bf16* X = reinterpret_cast<const __bf16*>(Xu);
  const __bf16* W = reinterpret_cast<const __bf16*>(Wu);
  __bf16* out = reinterpret_cast<__bf16*>(outu);

  const int tid  = threadIdx.x;
  const int w    = tid >> 5;
  const int lane = tid & 31;
  const int half = lane >> 4;   // 0: lanes 0-15, 1: lanes 16-31
  const int lr   = lane & 15;

  const int m0 = (blockIdx.x * 8 + w) * 32;  // wave token tile (32 rows)
  const int n0 = blockIdx.y * 64;            // output-column group

  __shared__ __attribute__((aligned(64))) __bf16 ldsW[2][64][32];  // [buf][j-n0][k-kk]

  v8f acc[2][4];
#pragma unroll
  for (int s = 0; s < 2; ++s)
#pragma unroll
    for (int t = 0; t < 4; ++t)
#pragma unroll
      for (int e = 0; e < 8; ++e) acc[s][t][e] = 0.f;

  // A fragment bases: rows m0+lr and m0+16+lr, per-lane K offsets (16-bit A layout)
  const __bf16* arow0 = X + (size_t)(m0 + lr) * DD + half * 8;
  const __bf16* arow1 = X + (size_t)(m0 + 16 + lr) * DD + half * 8;

  // Cooperative W staging indices: 256 threads x 8 elements = 64x32 slab
  const int wRow = tid >> 2;        // 0..63
  const int wSeg = (tid & 3) * 8;   // 0,8,16,24
  const __bf16* wsrc = W + (size_t)(n0 + wRow) * DD + wSeg;

  // Prologue: preload first slab into registers
  v8bf grw = *(const v8bf*)(wsrc);

  for (int kk = 0; kk < DD; kk += 32) {
    const int buf = (kk >> 5) & 1;
    // Commit current slab (loaded one iteration ago) to LDS
    *(v8bf*)&ldsW[buf][wRow][wSeg] = grw;
    // Issue next slab's global load early (uniform guard)
    const int kn = (kk + 32 < DD) ? (kk + 32) : kk;
    grw = *(const v8bf*)(wsrc + kn);
    __syncthreads();  // slab visible; also fences reads of this buffer 2 iters ago

    const v16bf a0 = cat8(*(const v8bf*)(arow0 + kk), *(const v8bf*)(arow0 + kk + 16));
    const v16bf a1 = cat8(*(const v8bf*)(arow1 + kk), *(const v8bf*)(arow1 + kk + 16));
#pragma unroll
    for (int t = 0; t < 4; ++t) {
      // B: column j = n0+t*16+lr; K in LDS row; lanes16-31 -> K+8 (table order)
      const v16bf bmat = cat8(*(const v8bf*)&ldsW[buf][t * 16 + lr][half * 8], *(const v8bf*)&ldsW[buf][t * 16 + lr][16 + half * 8]);
      acc[0][t] = __builtin_amdgcn_wmma_f32_16x16x32_bf16(
          false, a0, false, bmat, (short)0, acc[0][t], false, false);
      acc[1][t] = __builtin_amdgcn_wmma_f32_16x16x32_bf16(
          false, a1, false, bmat, (short)0, acc[1][t], false, false);
    }
  }

#pragma unroll
  for (int t = 0; t < 4; ++t) {
    const int j  = n0 + t * 16 + lr;   // output feature
    const int h  = j >> 6;
    const int dk = j & 63;
    const float bj = bias[j];
#pragma unroll
    for (int s = 0; s < 2; ++s) {
#pragma unroll
      for (int r = 0; r < 8; ++r) {
        const int token = m0 + s * 16 + half * 8 + r;  // C layout: M = half*8+r
        const int bidx  = token >> 11;                 // / L
        const int lrow  = token & (LL - 1);
        const __bf16 val = (__bf16)(acc[s][t][r] + bj);
        if (transposeV)
          out[(((size_t)bidx * HH + h) * DKK + dk) * LL + lrow] = val;
        else
          out[(((size_t)bidx * HH + h) * LL + lrow) * DKK + dk] = val;
      }
    }
  }
}

// ---------------------------------------------------------------------------
// Causal flash attention. Grid (L/64, B*H), block 128 (4 waves).
// Each wave: 16 query rows. 32-key tiles; K (32x64) and V^T (64x32) staged in
// double-buffered LDS shared by the 4 waves, software pipelined (next tile
// preloaded into registers a full iteration early). Two barriers per tile.
// ---------------------------------------------------------------------------
__global__ __launch_bounds__(128) void flash_kernel(const unsigned short* __restrict__ qhu,
                                                    const unsigned short* __restrict__ khu,
                                                    const unsigned short* __restrict__ vtu,
                                                    float* __restrict__ out) {
  const __bf16* qh  = reinterpret_cast<const __bf16*>(qhu);
  const __bf16* kh  = reinterpret_cast<const __bf16*>(khu);
  const __bf16* vhT = reinterpret_cast<const __bf16*>(vtu);

  const int tid  = threadIdx.x;
  const int w    = tid >> 5;
  const int lane = tid & 31;
  const int half = lane >> 4;
  const int lr   = lane & 15;

  const int bh = blockIdx.y;
  const int b  = bh >> 4;     // / H
  const int h  = bh & 15;
  const int q0 = blockIdx.x * 64 + w * 16;

  const __bf16* qbase = qh  + (size_t)bh * LL * DKK;
  const __bf16* kbase = kh  + (size_t)bh * LL * DKK;
  const __bf16* vbase = vhT + (size_t)bh * DKK * LL;

  // Q A-fragments: dk 0..31 (aq0) and 32..63 (aq1)
  const __bf16* qrow = qbase + (size_t)(q0 + lr) * DKK + half * 8;
  const v16bf aq0 = cat8(*(const v8bf*)(qrow),      *(const v8bf*)(qrow + 16));
  const v16bf aq1 = cat8(*(const v8bf*)(qrow + 32), *(const v8bf*)(qrow + 48));

  __shared__ __attribute__((aligned(64))) __bf16 ldsK[2][32][64];   // [buf][j-j0][dk]
  __shared__ __attribute__((aligned(64))) __bf16 ldsV[2][64][32];   // [buf][dk][j-j0]
  __shared__ __attribute__((aligned(64))) __bf16 lds_p[4][16][32];  // per-wave P

  // Cooperative staging indices (128 threads)
  const int kRow = tid >> 2;         // 0..31
  const int kSeg = (tid & 3) * 16;   // 0,16,32,48
  const int vRow = tid >> 1;         // 0..63
  const int vSeg = (tid & 1) * 16;   // 0,16
  const __bf16* ksrc = kbase + (size_t)kRow * DKK + kSeg;  // + j0*DKK per tile
  const __bf16* vsrc = vbase + (size_t)vRow * LL + vSeg;   // + j0 per tile

  v8f o[4];
  float mrow[8], lsum[8];
#pragma unroll
  for (int t = 0; t < 4; ++t)
#pragma unroll
    for (int e = 0; e < 8; ++e) o[t][e] = 0.f;
#pragma unroll
  for (int r = 0; r < 8; ++r) { mrow[r] = -1e30f; lsum[r] = 0.f; }

  // Uniform trip count per block (fully-masked tiles are exact no-ops)
  const int ntiles = 2 * blockIdx.x + 2;

  // Prologue: preload first K/V tile into registers
  v16bf grk = *(const v16bf*)(ksrc);
  v16bf grv = *(const v16bf*)(vsrc);

  for (int kt = 0; kt < ntiles; ++kt) {
    const int j0  = kt * 32;
    const int buf = kt & 1;

    // ---- commit current tile to LDS, issue next tile's loads early ----
    *(v16bf*)&ldsK[buf][kRow][kSeg] = grk;
    *(v16bf*)&ldsV[buf][vRow][vSeg] = grv;
    const int jn = (kt + 1 < ntiles) ? (j0 + 32) : j0;  // uniform guard
    grk = *(const v16bf*)(ksrc + (size_t)jn * DKK);
    grv = *(const v16bf*)(vsrc + jn);
    __syncthreads();  // tile visible

    // ---- S = Q K^T for 32 keys (two 16-col accumulators) ----
    v8f sA, sB;
#pragma unroll
    for (int e = 0; e < 8; ++e) { sA[e] = 0.f; sB[e] = 0.f; }
    {
      const v16bf bk0 = cat8(*(const v8bf*)&ldsK[buf][lr][half * 8], *(const v8bf*)&ldsK[buf][lr][16 + half * 8]);       // dk 0..31
      const v16bf bk1 = cat8(*(const v8bf*)&ldsK[buf][lr][32 + half * 8], *(const v8bf*)&ldsK[buf][lr][48 + half * 8]);  // dk 32..63
      sA = __builtin_amdgcn_wmma_f32_16x16x32_bf16(false, aq0, false, bk0, (short)0, sA, false, false);
      sA = __builtin_amdgcn_wmma_f32_16x16x32_bf16(false, aq1, false, bk1, (short)0, sA, false, false);
      const v16bf ck0 = cat8(*(const v8bf*)&ldsK[buf][16 + lr][half * 8], *(const v8bf*)&ldsK[buf][16 + lr][16 + half * 8]);
      const v16bf ck1 = cat8(*(const v8bf*)&ldsK[buf][16 + lr][32 + half * 8], *(const v8bf*)&ldsK[buf][16 + lr][48 + half * 8]);
      sB = __builtin_amdgcn_wmma_f32_16x16x32_bf16(false, aq0, false, ck0, (short)0, sB, false, false);
      sB = __builtin_amdgcn_wmma_f32_16x16x32_bf16(false, aq1, false, ck1, (short)0, sB, false, false);
    }

    // ---- mask + scale + online softmax (row reductions in 16-lane halves) ----
    float pA[8], pB[8];
#pragma unroll
    for (int r = 0; r < 8; ++r) {
      const int i  = q0 + half * 8 + r;
      const int jA = j0 + lr;
      const int jB = j0 + 16 + lr;
      float a  = (jA <= i) ? sA[r] * 0.125f : -1e30f;  // 1/sqrt(DK)=0.125
      float bb = (jB <= i) ? sB[r] * 0.125f : -1e30f;
      float tmax = fmaxf(a, bb);
#pragma unroll
      for (int m = 8; m >= 1; m >>= 1) tmax = fmaxf(tmax, __shfl_xor(tmax, m, 32));
      const float newm = fmaxf(mrow[r], tmax);
      const float ea = __expf(a - newm);
      const float eb = __expf(bb - newm);
      float tsum = ea + eb;
#pragma unroll
      for (int m = 8; m >= 1; m >>= 1) tsum += __shfl_xor(tsum, m, 32);
      const float corr = __expf(mrow[r] - newm);
      lsum[r] = lsum[r] * corr + tsum;
      mrow[r] = newm;
#pragma unroll
      for (int t = 0; t < 4; ++t) o[t][r] *= corr;
      pA[r] = ea; pB[r] = eb;
    }

    // ---- transpose P (C layout -> A layout) via per-wave LDS region ----
#pragma unroll
    for (int r = 0; r < 8; ++r) {
      lds_p[w][half * 8 + r][lr]      = (__bf16)pA[r];
      lds_p[w][half * 8 + r][16 + lr] = (__bf16)pB[r];
    }
    __syncthreads();  // P visible; also fences this iter's K reads
    const v16bf ap = cat8(*(const v8bf*)&lds_p[w][lr][half * 8],
                          *(const v8bf*)&lds_p[w][lr][16 + half * 8]);

    // ---- O += P V  (B operand from LDS V^T: contiguous in key dim) ----
#pragma unroll
    for (int t = 0; t < 4; ++t) {
      const v16bf bv = cat8(*(const v8bf*)&ldsV[buf][t * 16 + lr][half * 8], *(const v8bf*)&ldsV[buf][t * 16 + lr][16 + half * 8]);
      o[t] = __builtin_amdgcn_wmma_f32_16x16x32_bf16(
          false, ap, false, bv, (short)0, o[t], false, false);
    }
  }

  // ---- epilogue: out[b, i, h*64 + dk] = o / lsum ----
#pragma unroll
  for (int t = 0; t < 4; ++t) {
#pragma unroll
    for (int r = 0; r < 8; ++r) {
      const int i   = q0 + half * 8 + r;
      const int col = h * DKK + t * 16 + lr;
      out[((size_t)b * LL + i) * DD + col] = o[t][r] / lsum[r];
    }
  }
}

// ---------------------------------------------------------------------------
// Host launcher
// ---------------------------------------------------------------------------
extern "C" void kernel_launch(void* const* d_in, const int* in_sizes, int n_in,
                              void* d_out, int out_size, void* d_ws, size_t ws_size,
                              hipStream_t stream) {
  (void)in_sizes; (void)n_in; (void)out_size; (void)ws_size;

  const float* q    = (const float*)d_in[0];
  const float* k    = (const float*)d_in[1];
  const float* v    = (const float*)d_in[2];
  // d_in[3] = mask (causal, applied analytically in flash_kernel)
  const float* qn_g = (const float*)d_in[4];
  const float* qn_b = (const float*)d_in[5];
  const float* kn_g = (const float*)d_in[6];
  const float* kn_b = (const float*)d_in[7];
  const float* vn_g = (const float*)d_in[8];
  const float* vn_b = (const float*)d_in[9];
  const float* Wq   = (const float*)d_in[10];
  const float* bq   = (const float*)d_in[11];
  const float* Wk   = (const float*)d_in[12];
  const float* bk   = (const float*)d_in[13];
  const float* Wv   = (const float*)d_in[14];
  const float* bv   = (const float*)d_in[15];

  char* ws = (char*)d_ws;
  const size_t ACT  = (size_t)BB * LL * DD * sizeof(unsigned short); // 16 MB
  const size_t WSZ  = (size_t)DD * DD * sizeof(unsigned short);      //  2 MB
  unsigned short* xq  = (unsigned short*)(ws);
  unsigned short* xk  = (unsigned short*)(ws + ACT);
  unsigned short* xv  = (unsigned short*)(ws + 2 * ACT);
  unsigned short* qhp = (unsigned short*)(ws + 3 * ACT);
  unsigned short* khp = (unsigned short*)(ws + 4 * ACT);
  unsigned short* vtp = (unsigned short*)(ws + 5 * ACT);
  unsigned short* wqb = (unsigned short*)(ws + 6 * ACT);
  unsigned short* wkb = (unsigned short*)(ws + 6 * ACT + WSZ);
  unsigned short* wvb = (unsigned short*)(ws + 6 * ACT + 2 * WSZ);

  // 1) weights -> bf16
  const int nW = DD * DD;
  cast_kernel<<<(nW + 255) / 256, 256, 0, stream>>>(Wq, wqb, nW);
  cast_kernel<<<(nW + 255) / 256, 256, 0, stream>>>(Wk, wkb, nW);
  cast_kernel<<<(nW + 255) / 256, 256, 0, stream>>>(Wv, wvb, nW);

  // 2) layernorm + cast
  ln_cast_kernel<<<BB * LL, 256, 0, stream>>>(q, qn_g, qn_b, xq);
  ln_cast_kernel<<<BB * LL, 256, 0, stream>>>(k, kn_g, kn_b, xk);
  ln_cast_kernel<<<BB * LL, 256, 0, stream>>>(v, vn_g, vn_b, xv);

  // 3) projections (WMMA, double-buffered LDS weights). V stored transposed.
  dim3 pgrid(BB * LL / (32 * 8), DD / 64);
  proj_kernel<<<pgrid, 256, 0, stream>>>(xq, wqb, bq, qhp, 0);
  proj_kernel<<<pgrid, 256, 0, stream>>>(xk, wkb, bk, khp, 0);
  proj_kernel<<<pgrid, 256, 0, stream>>>(xv, wvb, bv, vtp, 1);

  // 4) causal flash attention (WMMA, double-buffered LDS K/V)
  dim3 agrid(LL / 64, BB * HH);
  flash_kernel<<<agrid, 128, 0, stream>>>(qhp, khp, vtp, (float*)d_out);
}
